// _ColNormLinear_74612171866401
// MI455X (gfx1250) — matrix operand order repaired and checked by reading, not compiled, not run on hardware
//
#include <hip/hip_runtime.h>
#include <hip/hip_bf16.h>
#include <stdint.h>

#define IN_F   4096
#define OUT_F  4096
#define NGRP_PER_COL 512   // OUT_F / 8
#define EPS_CLIP 1e-8f

typedef __attribute__((ext_vector_type(16))) __bf16 v16bf;
typedef __attribute__((ext_vector_type(8)))  float  v8f;
typedef int v4i __attribute__((vector_size(16)));

union FragAB { uint4 u[2]; v16bf v; };

// Async global->LDS (CDNA5). Guarded: fall back to sync loads if absent.
#if defined(__has_builtin)
#if __has_builtin(__builtin_amdgcn_global_load_async_to_lds_b128) && \
    __has_builtin(__builtin_amdgcn_s_wait_asynccnt)
#define USE_ASYNC 1
#endif
#endif
#ifndef USE_ASYNC
#define USE_ASYNC 0
#endif

// builtin signature wants v4i* in addrspace(1)/(3)
typedef __attribute__((address_space(1))) v4i* gv4p;
typedef __attribute__((address_space(3))) v4i* lv4p;

// round-to-nearest-even fp32 -> bf16 (bit pattern)
__device__ __forceinline__ unsigned short f2bf(float f) {
    unsigned int u = __builtin_bit_cast(unsigned int, f);
    u += 0x7FFFu + ((u >> 16) & 1u);
    return (unsigned short)(u >> 16);
}

// ---------------------------------------------------------------------------
// Kernel 1: x (fp32, row-major M x K) -> bf16 bits
// ---------------------------------------------------------------------------
__global__ void xconv_kernel(const float* __restrict__ x,
                             unsigned short* __restrict__ xb, int nchunks) {
    int i = blockIdx.x * blockDim.x + threadIdx.x;
    if (i >= nchunks) return;
    const float4* xv = (const float4*)x;
    float4 a = xv[2 * i];
    float4 b = xv[2 * i + 1];
    union { unsigned short h[8]; uint4 u; } p;
    p.h[0] = f2bf(a.x); p.h[1] = f2bf(a.y); p.h[2] = f2bf(a.z); p.h[3] = f2bf(a.w);
    p.h[4] = f2bf(b.x); p.h[5] = f2bf(b.y); p.h[6] = f2bf(b.z); p.h[7] = f2bf(b.w);
    ((uint4*)xb)[i] = p.u;
}

// ---------------------------------------------------------------------------
// Kernel 2: dequantize codebook weights into W (N rows, K cols) bf16.
// W[n][k] = centroids[labels[k*512 + n/8]][n%8] / max(h_scale[k],eps)
//            * col_std[k] + col_mean[k]
// ---------------------------------------------------------------------------
__global__ __launch_bounds__(256) void dequant_kernel(
        const float* __restrict__ centroids,
        const int*   __restrict__ labels,
        const float* __restrict__ col_mean,
        const float* __restrict__ col_std,
        const float* __restrict__ h_scale,
        unsigned short* __restrict__ Wb) {
    __shared__ unsigned short tile[64][72];   // [n_local][k_local], 144B row pitch
    const int bx = blockIdx.x;
    const int kt = bx & 63;
    const int nt = bx >> 6;
    const int k0 = kt * 64;
    const int n0 = nt * 64;
    const int n8_0 = nt * 8;
    const int t = threadIdx.x;

    for (int gi = t; gi < 512; gi += 256) {
        const int n8l = gi & 7;
        const int kl  = gi >> 3;
        const int k   = k0 + kl;
        const size_t g = (size_t)k * NGRP_PER_COL + (size_t)(n8_0 + n8l);
        const int lbl = labels[g];
        const float4* cp = (const float4*)(centroids + (size_t)lbl * 8);
        const float4 c0 = cp[0];
        const float4 c1 = cp[1];
        const float scale = col_std[k] / fmaxf(h_scale[k], EPS_CLIP);
        const float mean  = col_mean[k];
        const int nl = n8l * 8;
        tile[nl + 0][kl] = f2bf(fmaf(c0.x, scale, mean));
        tile[nl + 1][kl] = f2bf(fmaf(c0.y, scale, mean));
        tile[nl + 2][kl] = f2bf(fmaf(c0.z, scale, mean));
        tile[nl + 3][kl] = f2bf(fmaf(c0.w, scale, mean));
        tile[nl + 4][kl] = f2bf(fmaf(c1.x, scale, mean));
        tile[nl + 5][kl] = f2bf(fmaf(c1.y, scale, mean));
        tile[nl + 6][kl] = f2bf(fmaf(c1.z, scale, mean));
        tile[nl + 7][kl] = f2bf(fmaf(c1.w, scale, mean));
    }
    __syncthreads();
    const int r0 = t >> 3;
    const int cc = (t & 7) * 8;
    for (int r = r0; r < 64; r += 32) {
        uint4 v = *(const uint4*)&tile[r][cc];
        *(uint4*)&Wb[(size_t)(n0 + r) * IN_F + (size_t)(k0 + cc)] = v;
    }
}

// ---------------------------------------------------------------------------
// Kernel 3: C = Xb (M x K, bf16) * Wb^T (Wb is N x K, bf16) + bias, fp32 out.
// 128-thread block (4 waves), block tile 128M x 128N, wave tile 64x64
// (4x4 v_wmma_f32_16x16x32_bf16). Double-buffered LDS fed by async loads.
// K loop kept at unroll 1 (dynamic buffer toggle) to stay under 256 VGPRs.
// ---------------------------------------------------------------------------
#define LDS_PITCH 40                 // halves per row (80B): conflict-free frags
#define LDS_TILE  (128 * LDS_PITCH) // halves per buffer
__global__ __launch_bounds__(128) void gemm_kernel(
        const unsigned short* __restrict__ xb,
        const unsigned short* __restrict__ Wb,
        const float* __restrict__ bias,
        float* __restrict__ out, int M) {
    __shared__ unsigned short As[2 * LDS_TILE];
    __shared__ unsigned short Bs[2 * LDS_TILE];

    const int tid  = threadIdx.x;               // 0..127
    const int bm   = blockIdx.y * 128;
    const int bn   = blockIdx.x * 128;
    const int wave = tid >> 5;
    const int lane = tid & 31;
    const int wm = (wave >> 1) * 64;            // wave M origin: 0 / 64
    const int wn = (wave & 1) * 64;             // wave N origin: 0 / 64
    const int lane16   = lane & 15;
    const int laneHalf = lane >> 4;

    v8f acc[4][4];
#pragma unroll
    for (int i = 0; i < 4; ++i)
#pragma unroll
        for (int j = 0; j < 4; ++j) acc[i][j] = (v8f)0.0f;

    // each thread owns one 64B row-chunk of each operand tile
    const unsigned short* gA = xb + (size_t)(bm + tid) * IN_F;
    const unsigned short* gB = Wb + (size_t)(bn + tid) * IN_F;

    // per-thread LDS bases (element offsets)
    unsigned short* const sA = As + tid * LDS_PITCH;   // store side
    unsigned short* const sB = Bs + tid * LDS_PITCH;
    const int aRow = (wm + lane16) * LDS_PITCH + laneHalf * 8;    // frag side
    const int bRow = (wn + lane16) * LDS_PITCH + laneHalf * 8;

    auto issue = [&](unsigned bufOff, int kb) {
#if USE_ASYNC
#pragma unroll
        for (int q = 0; q < 4; ++q) {
            __builtin_amdgcn_global_load_async_to_lds_b128(
                (gv4p)(uintptr_t)(gA + kb + q * 8),
                (lv4p)(uintptr_t)(sA + bufOff + q * 8), 0, 0);
            __builtin_amdgcn_global_load_async_to_lds_b128(
                (gv4p)(uintptr_t)(gB + kb + q * 8),
                (lv4p)(uintptr_t)(sB + bufOff + q * 8), 0, 0);
        }
#else
#pragma unroll
        for (int q = 0; q < 4; ++q) {
            uint4 va = *(const uint4*)(gA + kb + q * 8);
            uint4 vb = *(const uint4*)(gB + kb + q * 8);
            *(uint4*)(sA + bufOff + q * 8) = va;
            *(uint4*)(sB + bufOff + q * 8) = vb;
        }
#endif
    };

    auto compute = [&](unsigned bufOff) {
        // A frag: lane<16 -> M=lane16, K {0..7}U{16..23}; lane>=16 -> K {8..15}U{24..31}
        FragAB a[4];
#pragma unroll
        for (int i = 0; i < 4; ++i) {
            const unsigned short* p = As + bufOff + aRow + i * (16 * LDS_PITCH);
            a[i].u[0] = *(const uint4*)p;
            a[i].u[1] = *(const uint4*)(p + 16);
        }
        // B frag: lane<16 -> N=lane16, K {0..7}U{16..23}; lane>=16 -> K {8..15}U{24..31}
        FragAB b[4];
#pragma unroll
        for (int j = 0; j < 4; ++j) {
            const unsigned short* p = Bs + bufOff + bRow + j * (16 * LDS_PITCH);
            b[j].u[0] = *(const uint4*)p;
            b[j].u[1] = *(const uint4*)(p + 16);
        }
#pragma unroll
        for (int i = 0; i < 4; ++i)
#pragma unroll
            for (int j = 0; j < 4; ++j)
                acc[i][j] = __builtin_amdgcn_wmma_f32_16x16x32_bf16(
                    false, a[i].v, false, b[j].v, (short)0, acc[i][j],
                    false, false);
    };

    const int ntiles = IN_F / 32;
    unsigned cur = 0;

    issue(0, 0);
#pragma unroll 1
    for (int it = 0; it < ntiles - 1; ++it) {
        issue(cur ^ LDS_TILE, (it + 1) * 32);
#if USE_ASYNC
        __builtin_amdgcn_s_wait_asynccnt(8);   // current tile's 8 done (in-order)
#endif
        __syncthreads();     // all waves' current-tile data visible
        compute(cur);
        __syncthreads();     // protect buffer the next iteration overwrites
        cur ^= LDS_TILE;
    }
#if USE_ASYNC
    __builtin_amdgcn_s_wait_asynccnt(0);
#endif
    __syncthreads();
    compute(cur);

    // Epilogue: C VGPR r -> (M = r + 8*laneHalf, N = lane16) per 16x16 tile
#pragma unroll
    for (int j = 0; j < 4; ++j) {
        const int n = bn + wn + j * 16 + lane16;
        const float bv = bias[n];
#pragma unroll
        for (int i = 0; i < 4; ++i) {
            const int mb = bm + wm + i * 16 + laneHalf * 8;
            float* o = out + (size_t)mb * OUT_F + n;
#pragma unroll
            for (int r = 0; r < 8; ++r)
                o[(size_t)r * OUT_F] = acc[i][j][r] + bv;
        }
    }
}

// ---------------------------------------------------------------------------
extern "C" void kernel_launch(void* const* d_in, const int* in_sizes, int n_in,
                              void* d_out, int out_size, void* d_ws, size_t ws_size,
                              hipStream_t stream) {
    const float* x         = (const float*)d_in[0];
    const float* centroids = (const float*)d_in[1];
    const int*   labels    = (const int*)d_in[2];
    const float* col_mean  = (const float*)d_in[3];
    const float* col_std   = (const float*)d_in[4];
    const float* h_scale   = (const float*)d_in[5];
    const float* bias      = (const float*)d_in[6];
    float* out = (float*)d_out;

    const int M = in_sizes[0] / IN_F;    // 2048

    unsigned short* xb = (unsigned short*)d_ws;                 // M*K bf16
    unsigned short* Wb = xb + (size_t)M * IN_F;                 // N*K bf16

    const int nchunks = (M * IN_F) / 8;
    xconv_kernel<<<(nchunks + 255) / 256, 256, 0, stream>>>(x, xb, nchunks);

    dequant_kernel<<<(OUT_F / 64) * (IN_F / 64), 256, 0, stream>>>(
        centroids, labels, col_mean, col_std, h_scale, Wb);

    dim3 grid(OUT_F / 128, M / 128);
    gemm_kernel<<<grid, 128, 0, stream>>>(xb, Wb, bias, out, M);
}
